// DCN_layer_rgb_56358560858419
// MI455X (gfx1250) — matrix operand order repaired and checked by reading, not compiled, not run on hardware
//
#include <hip/hip_runtime.h>
#include <hip/hip_bf16.h>
#include <math.h>

typedef _Float16 f16;
typedef __attribute__((ext_vector_type(16))) _Float16 v16h;
typedef __attribute__((ext_vector_type(8)))  _Float16 v8h;
typedef __attribute__((ext_vector_type(8)))  float    v8f;

#define HH 128
#define WW 128
#define HW 16384   // HH*WW

// ---------------------------------------------------------------------------
// Prep: fvec[b][o] = sum_j c1_w[o,j] * fea[b,j]
// ---------------------------------------------------------------------------
__global__ __launch_bounds__(64) void k_fvec(const float* __restrict__ c1w,
                                             const float* __restrict__ fea,
                                             float* __restrict__ fvec)
{
    const int b = blockIdx.x, o = threadIdx.x;
    const float* crow = c1w + o * 256;
    const float* frow = fea + b * 256;
    float s = 0.f;
    for (int j = 0; j < 256; ++j) s += crow[j] * frow[j];
    fvec[b * 64 + o] = s;
}

// ---------------------------------------------------------------------------
// Prep: wdyn16[b][o][k = p*64+c] = f16( weight[o,c,p] * fvec[b][c] )
// (K reordered to p*64+c so the gather can write contiguous channels)
// ---------------------------------------------------------------------------
__global__ __launch_bounds__(64) void k_wdyn(const float* __restrict__ weight,
                                             const float* __restrict__ fvec,
                                             f16* __restrict__ wdyn)
{
    const int b = blockIdx.x >> 6, o = blockIdx.x & 63;
    const int c = threadIdx.x;
    const float f = fvec[b * 64 + c];
    f16* row = wdyn + (size_t)(b * 64 + o) * 576;
    const float* wrow = weight + (o * 64 + c) * 9;
    for (int p = 0; p < 9; ++p) row[p * 64 + c] = (f16)(wrow[p] * f);
}

// ---------------------------------------------------------------------------
// Prep: c2_w f32 -> f16
// ---------------------------------------------------------------------------
__global__ __launch_bounds__(256) void k_c2cvt(const float* __restrict__ c2w,
                                               f16* __restrict__ c2h)
{
    const int i = blockIdx.x * 256 + threadIdx.x;
    if (i < 4096) c2h[i] = (f16)c2w[i];
}

// ---------------------------------------------------------------------------
// Offset conv: om = conv3x3(inter, com_w)+com_b -> dy/dx/mask split.
// One thread per (b, ch, pixel); ch uniform within a block -> weights become
// scalar (s_load) operands.
// ---------------------------------------------------------------------------
__global__ __launch_bounds__(256) void k_offsets(const float* __restrict__ inter,
                                                 const float* __restrict__ comw,
                                                 const float* __restrict__ comb,
                                                 float* __restrict__ dyo,
                                                 float* __restrict__ dxo,
                                                 float* __restrict__ msko)
{
    const int gid = blockIdx.x * 256 + threadIdx.x;
    const int pix = gid & (HW - 1);
    const int bc  = gid >> 14;           // b*27 + ch
    const int b   = bc / 27;
    const int ch  = bc - 27 * b;
    const int y = pix >> 7, x = pix & 127;

    float acc = comb[ch];
    const float* wbase = comw + ch * 64 * 9;
    const float* ibase = inter + (size_t)b * 64 * HW;
    for (int ky = 0; ky < 3; ++ky) {
        const int yy = y + ky - 1;
        if (yy < 0 || yy >= HH) continue;
        for (int kx = 0; kx < 3; ++kx) {
            const int xx = x + kx - 1;
            if (xx < 0 || xx >= WW) continue;
            const int t = ky * 3 + kx;
            const float* ip = ibase + yy * WW + xx;
            for (int c = 0; c < 64; ++c)
                acc += ip[(size_t)c * HW] * wbase[c * 9 + t];
        }
    }
    if (ch < 18) {
        float* dst = ((ch & 1) == 0) ? dyo : dxo;
        dst[(size_t)(b * 9 + (ch >> 1)) * HW + pix] = acc;
    } else {
        msko[(size_t)(b * 9 + (ch - 18)) * HW + pix] = 1.f / (1.f + expf(-acc));
    }
}

// ---------------------------------------------------------------------------
// Main fused kernel: per 16-pixel tile
//   val(576x16) bilinear gather -> LDS (f16)
//   out(64x16)  = wdyn(64x576) x val        [18 WMMA / wave]
//   fin(64x16)  = c2 (64x64)  x (out+bias)  [ 2 WMMA / wave]
// 4 waves = 4 M-tiles; f32 accumulation throughout.
// ---------------------------------------------------------------------------
__global__ __launch_bounds__(128) void k_main(const float* __restrict__ xin,
                                              const f16* __restrict__ wdyn,
                                              const f16* __restrict__ c2h,
                                              const float* __restrict__ bias,
                                              const float* __restrict__ dyo,
                                              const float* __restrict__ dxo,
                                              const float* __restrict__ msko,
                                              float* __restrict__ out)
{
    __shared__ f16 val_lds[16][584];  // [pixel][k], k = p*64+c, padded row
    __shared__ f16 out_lds[16][72];   // [pixel][o], padded row

    const int blk  = blockIdx.x;
    const int b    = blk >> 10;
    const int tile = blk & 1023;
    const int y    = tile >> 3;
    const int xb   = (tile & 7) << 4;
    const int tid  = threadIdx.x;
    const int wave = tid >> 5;
    const int lane = tid & 31;
    const int m    = lane & 15;       // row (A) / pixel (B) within tile
    const int g    = lane >> 4;       // lane half-group

    // ---- build val in LDS: 144 (pixel, tap) jobs over 128 threads ----
    const float* xbase = xin + (size_t)b * 64 * HW;
    for (int i = tid; i < 144; i += 128) {
        const int pix = i & 15;
        const int p   = i >> 4;
        const int x   = xb + pix;
        const int oidx = (b * 9 + p) * HW + y * WW + x;
        const float dyv = dyo[oidx];
        const float dxv = dxo[oidx];
        const float mv  = msko[oidx];
        const int ky = p / 3, kx = p - 3 * (p / 3);
        const float ys = (float)(y - 1 + ky) + dyv;
        const float xs = (float)(x - 1 + kx) + dxv;
        const float y0f = floorf(ys), x0f = floorf(xs);
        const float ly = ys - y0f, lx = xs - x0f;
        const int y0 = (int)y0f, x0 = (int)x0f;
        const int y1 = y0 + 1, x1 = x0 + 1;
        const float vy0 = (y0 >= 0 && y0 < HH) ? 1.f : 0.f;
        const float vy1 = (y1 >= 0 && y1 < HH) ? 1.f : 0.f;
        const float vx0 = (x0 >= 0 && x0 < WW) ? 1.f : 0.f;
        const float vx1 = (x1 >= 0 && x1 < WW) ? 1.f : 0.f;
        const float w00 = (1.f - ly) * (1.f - lx) * vy0 * vx0 * mv;
        const float w01 = (1.f - ly) * lx * vy0 * vx1 * mv;
        const float w10 = ly * (1.f - lx) * vy1 * vx0 * mv;
        const float w11 = ly * lx * vy1 * vx1 * mv;
        const int cy0 = min(max(y0, 0), HH - 1), cy1 = min(max(y1, 0), HH - 1);
        const int cx0 = min(max(x0, 0), WW - 1), cx1 = min(max(x1, 0), WW - 1);
        const int i00 = cy0 * WW + cx0, i01 = cy0 * WW + cx1;
        const int i10 = cy1 * WW + cx0, i11 = cy1 * WW + cx1;
        const int kbase = p * 64;
        for (int c = 0; c < 64; c += 8) {
            v8h tmp;
#pragma unroll
            for (int u = 0; u < 8; ++u) {
                const float* cb = xbase + (size_t)(c + u) * HW;
                const float v = cb[i00] * w00 + cb[i01] * w01 +
                                cb[i10] * w10 + cb[i11] * w11;
                tmp[u] = (f16)v;
            }
            *(v8h*)(&val_lds[pix][kbase + c]) = tmp;  // 16B aligned
        }
    }
    __syncthreads();

    // ---- stage 1: out = wdyn x val, K=576 in 18 steps of 32 ----
    v8f acc = {};
    const f16* arow = wdyn + ((size_t)(b * 64) + wave * 16 + m) * 576;
#pragma unroll
    for (int kk = 0; kk < 18; ++kk) {
        const int kA = kk * 32 + g * 8;          // A layout: [k..k+7],[k+16..k+23]
        const v8h alo = *(const v8h*)(arow + kA);
        const v8h ahi = *(const v8h*)(arow + kA + 16);
        const int kB = kk * 32 + g * 8;          // B layout: [k..k+7],[k+16..k+23]
        const v8h blo = *(const v8h*)(&val_lds[m][kB]);
        const v8h bhi = *(const v8h*)(&val_lds[m][kB + 16]);
        v16h a, bb;
#pragma unroll
        for (int u = 0; u < 8; ++u) {
            a[u] = alo[u]; a[u + 8] = ahi[u];
            bb[u] = blo[u]; bb[u + 8] = bhi[u];
        }
        acc = __builtin_amdgcn_wmma_f32_16x16x32_f16(
            false, a, false, bb, (short)0, acc, false, false);
    }

    // ---- restage (out + bias) as f16 B-operand for the c2 projection ----
    {
        v8h pk;
#pragma unroll
        for (int r = 0; r < 8; ++r)
            pk[r] = (f16)(acc[r] + bias[wave * 16 + g * 8 + r]);
        *(v8h*)(&out_lds[m][wave * 16 + g * 8]) = pk;  // D rows -> contiguous K
    }
    __syncthreads();

    // ---- stage 2: fin = c2 x out, K=64 in 2 steps ----
    v8f acc2 = {};
    const f16* a2 = c2h + (wave * 16 + m) * 64;
#pragma unroll
    for (int kk = 0; kk < 2; ++kk) {
        const int kA = kk * 32 + g * 8;
        const v8h alo = *(const v8h*)(a2 + kA);
        const v8h ahi = *(const v8h*)(a2 + kA + 16);
        const int kB = kk * 32 + g * 8;
        const v8h blo = *(const v8h*)(&out_lds[m][kB]);
        const v8h bhi = *(const v8h*)(&out_lds[m][kB + 16]);
        v16h a, bb;
#pragma unroll
        for (int u = 0; u < 8; ++u) {
            a[u] = alo[u]; a[u + 8] = ahi[u];
            bb[u] = blo[u]; bb[u + 8] = bhi[u];
        }
        acc2 = __builtin_amdgcn_wmma_f32_16x16x32_f16(
            false, a, false, bb, (short)0, acc2, false, false);
    }

    // ---- store: element (o2 = wave*16 + g*8 + r, pixel = m) ----
    float* orow = out + ((size_t)(b * 64) + wave * 16 + g * 8) * HW + y * WW + xb + m;
#pragma unroll
    for (int r = 0; r < 8; ++r) orow[(size_t)r * HW] = acc2[r];
}

// ---------------------------------------------------------------------------
extern "C" void kernel_launch(void* const* d_in, const int* in_sizes, int n_in,
                              void* d_out, int out_size, void* d_ws, size_t ws_size,
                              hipStream_t stream)
{
    const float* input_feat = (const float*)d_in[0];
    const float* inter      = (const float*)d_in[1];
    const float* fea        = (const float*)d_in[2];
    const float* weight     = (const float*)d_in[3];
    const float* bias       = (const float*)d_in[4];
    const float* com_w      = (const float*)d_in[5];
    const float* com_b      = (const float*)d_in[6];
    const float* c1_w       = (const float*)d_in[7];
    const float* c2_w       = (const float*)d_in[8];
    float* out = (float*)d_out;

    char* ws = (char*)d_ws;
    float* fvec = (float*)(ws);                          //   1,024 B
    f16*   wdyn = (f16*)(ws + 1024);                     // 294,912 B
    f16*   c2h  = (f16*)(ws + 1024 + 294912);            //   8,192 B
    float* dyo  = (float*)(ws + 304128);                 // 2,359,296 B
    float* dxo  = (float*)(ws + 304128 + 2359296);       // 2,359,296 B
    float* msko = (float*)(ws + 304128 + 2 * 2359296);   // 2,359,296 B

    k_fvec   <<<4,    64,  0, stream>>>(c1_w, fea, fvec);
    k_wdyn   <<<256,  64,  0, stream>>>(weight, fvec, wdyn);
    k_c2cvt  <<<16,   256, 0, stream>>>(c2_w, c2h);
    k_offsets<<<6912, 256, 0, stream>>>(inter, com_w, com_b, dyo, dxo, msko);
    k_main   <<<4096, 128, 0, stream>>>(input_feat, wdyn, c2h, bias,
                                        dyo, dxo, msko, out);
}
